// GCONVCheb_24026047054158
// MI455X (gfx1250) — matrix operand order repaired and checked by reading, not compiled, not run on hardware
//
#include <hip/hip_runtime.h>
#include <hip/hip_bf16.h>
#include <math.h>

// Problem constants (match reference)
#define N_NODES 100000
#define N_EDGES 3200000
#define EMB     10
#define OUTC    16
#define KCHEB   5
#define DD      8
#define TGT     3

typedef __attribute__((ext_vector_type(16))) _Float16 v16h;
typedef __attribute__((ext_vector_type(8)))  float    v8f;

static __device__ __forceinline__ v8f wmma_f32_16x16x32_f16(v16h a, v16h b, v8f c) {
  // 8 args: (neg_a, A, neg_b, B, c_mod, C, reuse_a, reuse_b)
  return __builtin_amdgcn_wmma_f32_16x16x32_f16(false, a, false, b, (short)0, c, false, false);
}

// ---------------------------------------------------------------------------
// Graph normalization: deg -> dinv -> edge weight w = -dinv[row]*dinv[col]
// ---------------------------------------------------------------------------
__global__ void k_degree(const int* __restrict__ ei, float* __restrict__ deg) {
  int e = blockIdx.x * blockDim.x + threadIdx.x;           // exact grid
  int r = ei[e], c = ei[N_EDGES + e];
  if (r != c) atomicAdd(deg + r, 1.0f);
}

__global__ void k_dinv(float* __restrict__ deg) {
  int n = blockIdx.x * blockDim.x + threadIdx.x;
  if (n >= N_NODES) return;
  float d = deg[n];
  deg[n] = (d > 0.f) ? rsqrtf(fmaxf(d, 1.f)) : 0.f;        // in-place dinv
}

__global__ void k_edgew(const int* __restrict__ ei, const float* __restrict__ dinv,
                        float* __restrict__ w) {
  int e = blockIdx.x * blockDim.x + threadIdx.x;           // exact grid
  int r = ei[e], c = ei[N_EDGES + e];
  w[e] = (r != c) ? (-dinv[r] * dinv[c]) : 0.f;
}

// ---------------------------------------------------------------------------
// Chebyshev propagation: acc init (0 or -T_{k-2}) then scatter 2*w*T_{k-1}[row]
// Accumulator is 4 MB -> L2 resident on MI455X (192 MB L2); f32 atomics.
// ---------------------------------------------------------------------------
__global__ void k_initT(float* __restrict__ dst, const float* __restrict__ prev2) {
  int i = blockIdx.x * blockDim.x + threadIdx.x;
  if (i >= N_NODES * EMB) return;
  dst[i] = prev2 ? -prev2[i] : 0.f;
}

__global__ void k_scatter(const int* __restrict__ ei, const float* __restrict__ w,
                          const float* __restrict__ src, float* __restrict__ acc,
                          float scale) {
  int e = blockIdx.x * blockDim.x + threadIdx.x;           // exact grid
  float we = w[e] * scale;
  if (we == 0.f) return;                                   // self-loops / isolated
  int r = ei[e], c = ei[N_EDGES + e];
  const float* s = src + (size_t)r * EMB;
  float* a = acc + (size_t)c * EMB;
#pragma unroll
  for (int j = 0; j < EMB; ++j) atomicAdd(a + j, we * s[j]);
}

// ---------------------------------------------------------------------------
// Node transform + GRU gates (H0=0 simplification):
//   z = sigmoid(sum_k T_k @ Wxz[k] + bxz + bhz)
//   t = tanh   (sum_k T_k @ Wxh[k] + bxh + bhh)
//   h = relu((1-z)*t)  -> stored as f16 to halve edge-gather traffic
// ---------------------------------------------------------------------------
__global__ void k_node(const float* __restrict__ T0, const float* __restrict__ T1,
                       const float* __restrict__ T2, const float* __restrict__ T3,
                       const float* __restrict__ T4,
                       const float* __restrict__ Wxz, const float* __restrict__ bxz,
                       const float* __restrict__ bhz,
                       const float* __restrict__ Wxh, const float* __restrict__ bxh,
                       const float* __restrict__ bhh,
                       _Float16* __restrict__ h16) {
  __shared__ float sWz[KCHEB * EMB * OUTC];
  __shared__ float sWh[KCHEB * EMB * OUTC];
  __shared__ float sbz[OUTC];
  __shared__ float sbh[OUTC];
  int tid = threadIdx.x;
  for (int i = tid; i < KCHEB * EMB * OUTC; i += blockDim.x) { sWz[i] = Wxz[i]; sWh[i] = Wxh[i]; }
  if (tid < OUTC) { sbz[tid] = bxz[tid] + bhz[tid]; sbh[tid] = bxh[tid] + bhh[tid]; }
  __syncthreads();

  int n = blockIdx.x * blockDim.x + tid;
  if (n >= N_NODES) return;
  const float* Ts[KCHEB] = {T0, T1, T2, T3, T4};
  float az[OUTC], ah[OUTC];
#pragma unroll
  for (int o = 0; o < OUTC; ++o) { az[o] = 0.f; ah[o] = 0.f; }
#pragma unroll
  for (int k = 0; k < KCHEB; ++k) {
    const float* t = Ts[k] + (size_t)n * EMB;
#pragma unroll
    for (int i = 0; i < EMB; ++i) {
      float tv = t[i];
      const float* wz = &sWz[(k * EMB + i) * OUTC];
      const float* wh = &sWh[(k * EMB + i) * OUTC];
#pragma unroll
      for (int o = 0; o < OUTC; ++o) { az[o] += tv * wz[o]; ah[o] += tv * wh[o]; }
    }
  }
#pragma unroll
  for (int o = 0; o < OUTC; ++o) {
    float z  = 1.f / (1.f + expf(-(az[o] + sbz[o])));
    float ht = tanhf(ah[o] + sbh[o]);
    float hv = fmaxf((1.f - z) * ht, 0.f);
    h16[(size_t)n * OUTC + o] = (_Float16)hv;
  }
}

// ---------------------------------------------------------------------------
// Pack transposed MLP weights into WMMA A-operand register layout (f16).
// A 16x32 f16 layout (contiguous k): lane<16: slots 0..15 -> K=0..15,
//                     lane>=16: slots 0..15 -> K=16..31; row M = lane&15.
// A[M=out_channel][K=in_dim] = W[K][M] (W stored row-major [Kdim, Ndim]).
// 6 matrices: W0^T (32x8), Wmid[0..3]^T (8x8), Wlast^T (8x3).
// ---------------------------------------------------------------------------
__global__ void k_pack(const float* __restrict__ W0, const float* __restrict__ Wmid,
                       const float* __restrict__ Wlast, _Float16* __restrict__ packW) {
  int t = blockIdx.x * blockDim.x + threadIdx.x;
  if (t >= 6 * 32 * 16) return;
  int mat = t >> 9;
  int l   = (t >> 4) & 31;
  int s   = t & 15;
  const float* W; int Kd, Nd;
  if (mat == 0)      { W = W0;                         Kd = 2 * OUTC; Nd = DD; }
  else if (mat <= 4) { W = Wmid + (mat - 1) * DD * DD; Kd = DD;       Nd = DD; }
  else               { W = Wlast;                      Kd = DD;       Nd = TGT; }
  int kb = (l < 16) ? 0 : 16;
  int k  = kb + s;
  int m  = l & 15;
  float v = (k < Kd && m < Nd) ? W[k * Nd + m] : 0.f;
  packW[t] = (_Float16)v;   // packW[mat*512 + l*16 + s]
}

// ---------------------------------------------------------------------------
// Edge MLP via WMMA, one wave = 16 edges, computed transposed (yT = WT @ featT).
// B operand (32x16 f16): lane n (<16) = edge n home features h16[home][0..15]
// (one contiguous 32B v16h), lane n+16 = away features (K=16..31).
// D (channels x edges) feeds next layer's B after bias+relu+cvt.
//
// Uniform control flow throughout: lanes>=16 and slots 8..15 carry junk after
// the bias add, but those B slots map to K>=8 where the next packed A matrix
// is all zeros, so they contribute nothing. Only the final store is predicated.
// ---------------------------------------------------------------------------
__global__ void k_mlp(const int* __restrict__ ei, const _Float16* __restrict__ h16,
                      const _Float16* __restrict__ packW,
                      const float* __restrict__ b0, const float* __restrict__ bmid,
                      const float* __restrict__ blast, float* __restrict__ out) {
  int wave = (int)((blockIdx.x * blockDim.x + threadIdx.x) >> 5);
  int lane = threadIdx.x & 31;
  int n    = lane & 15;
  int sel  = lane >> 4;
  int e    = wave * 16 + n;                     // E is a multiple of 16, exact grid

  // Hoist all biases up-front: uniform addresses -> batched scalar loads.
  float bb0[DD], bbm[4][DD], bl[TGT];
#pragma unroll
  for (int i = 0; i < DD; ++i) bb0[i] = b0[i];
#pragma unroll
  for (int l = 0; l < 4; ++l)
#pragma unroll
    for (int i = 0; i < DD; ++i) bbm[l][i] = bmid[l * DD + i];
#pragma unroll
  for (int i = 0; i < TGT; ++i) bl[i] = blast[i];

  int idx = sel ? ei[N_EDGES + e] : ei[e];      // away : home endpoint
  v16h featB = *(const v16h*)(h16 + (size_t)idx * OUTC);   // 32B aligned

  // layer 0: [8x32] @ [32x16] (padded to 16x32 x 32x16)
  v16h a = *(const v16h*)(packW + lane * 16);
  v8f zc = {};
  v8f d = wmma_f32_16x16x32_f16(a, featB, zc);

  v16h y = {};                                  // slots 8..15 stay zero
#pragma unroll
  for (int i = 0; i < DD; ++i) y[i] = (_Float16)fmaxf(d[i] + bb0[i], 0.f);

#pragma unroll
  for (int l = 0; l < 4; ++l) {                 // 4 mid layers 8->8
    a = *(const v16h*)(packW + (1 + l) * 512 + lane * 16);
    v8f z2 = {};
    d = wmma_f32_16x16x32_f16(a, y, z2);
#pragma unroll
    for (int i = 0; i < DD; ++i) y[i] = (_Float16)fmaxf(d[i] + bbm[l][i], 0.f);
  }

  a = *(const v16h*)(packW + 5 * 512 + lane * 16);   // last layer 8->3
  v8f z3 = {};
  d = wmma_f32_16x16x32_f16(a, y, z3);

  // relu + softmax over 3 channels (computed by all lanes; stored by lanes<16)
  float v0 = fmaxf(d[0] + bl[0], 0.f);
  float v1 = fmaxf(d[1] + bl[1], 0.f);
  float v2 = fmaxf(d[2] + bl[2], 0.f);
  float mx = fmaxf(v0, fmaxf(v1, v2));
  float e0 = expf(v0 - mx), e1 = expf(v1 - mx), e2 = expf(v2 - mx);
  float inv = 1.f / (e0 + e1 + e2);
  if (sel == 0) {
    float* o = out + (size_t)e * TGT;
    o[0] = e0 * inv; o[1] = e1 * inv; o[2] = e2 * inv;
  }
}

// ---------------------------------------------------------------------------
extern "C" void kernel_launch(void* const* d_in, const int* in_sizes, int n_in,
                              void* d_out, int out_size, void* d_ws, size_t ws_size,
                              hipStream_t stream) {
  (void)in_sizes; (void)n_in; (void)out_size; (void)ws_size;
  const int*   ei    = (const int*)  d_in[0];
  const float* emb   = (const float*)d_in[1];
  const float* Wxz   = (const float*)d_in[2];
  const float* bxz   = (const float*)d_in[3];
  const float* bhz   = (const float*)d_in[5];
  const float* Wxh   = (const float*)d_in[10];
  const float* bxh   = (const float*)d_in[11];
  const float* bhh   = (const float*)d_in[13];
  const float* W0    = (const float*)d_in[14];
  const float* b0    = (const float*)d_in[15];
  const float* Wmid  = (const float*)d_in[16];
  const float* bmid  = (const float*)d_in[17];
  const float* Wlast = (const float*)d_in[18];
  const float* blast = (const float*)d_in[19];
  float* out = (float*)d_out;

  // Workspace carve-out (256B-aligned regions, ~32.4 MB total)
  char* ws = (char*)d_ws;
  size_t off = 0;
  auto carve = [&](size_t bytes) -> char* {
    char* p = ws + off;
    off += (bytes + 255) & ~(size_t)255;
    return p;
  };
  float*    w     = (float*)carve((size_t)N_EDGES * 4);
  float*    T1    = (float*)carve((size_t)N_NODES * EMB * 4);
  float*    T2    = (float*)carve((size_t)N_NODES * EMB * 4);
  float*    T3    = (float*)carve((size_t)N_NODES * EMB * 4);
  float*    T4    = (float*)carve((size_t)N_NODES * EMB * 4);
  float*    deg   = (float*)carve((size_t)N_NODES * 4);
  _Float16* h16   = (_Float16*)carve((size_t)N_NODES * OUTC * 2);
  _Float16* packW = (_Float16*)carve((size_t)6 * 512 * 2);

  const int EB = N_EDGES / 256;            // 12500, exact
  const int NB = (N_NODES + 255) / 256;    // 391
  const int TB = (N_NODES * EMB + 255) / 256;

  // Normalization
  hipMemsetAsync(deg, 0, (size_t)N_NODES * 4, stream);
  k_degree<<<EB, 256, 0, stream>>>(ei, deg);
  k_dinv  <<<NB, 256, 0, stream>>>(deg);
  k_edgew <<<EB, 256, 0, stream>>>(ei, deg, w);

  // Chebyshev recursion: T1 = P x; Tk = 2 P T_{k-1} - T_{k-2}
  k_initT  <<<TB, 256, 0, stream>>>(T1, nullptr);
  k_scatter<<<EB, 256, 0, stream>>>(ei, w, emb, T1, 1.f);
  k_initT  <<<TB, 256, 0, stream>>>(T2, emb);
  k_scatter<<<EB, 256, 0, stream>>>(ei, w, T1, T2, 2.f);
  k_initT  <<<TB, 256, 0, stream>>>(T3, T1);
  k_scatter<<<EB, 256, 0, stream>>>(ei, w, T2, T3, 2.f);
  k_initT  <<<TB, 256, 0, stream>>>(T4, T2);
  k_scatter<<<EB, 256, 0, stream>>>(ei, w, T3, T4, 2.f);

  // Gates + node features (f16)
  k_node<<<NB, 256, 0, stream>>>(emb, T1, T2, T3, T4,
                                 Wxz, bxz, bhz, Wxh, bxh, bhh, h16);

  // Pack MLP weights into WMMA A-operand layout, then run edge MLP
  k_pack<<<(6 * 32 * 16 + 255) / 256, 256, 0, stream>>>(W0, Wmid, Wlast, packW);
  k_mlp<<<(N_EDGES / 16) / 8, 256, 0, stream>>>(ei, h16, packW, b0, bmid, blast, out);
}
